// cop_attn_head_88802743812778
// MI455X (gfx1250) — matrix operand order repaired and checked by reading, not compiled, not run on hardware
//
#include <hip/hip_runtime.h>
#include <hip/hip_bf16.h>

typedef __attribute__((ext_vector_type(16))) _Float16 v16h;
typedef __attribute__((ext_vector_type(8)))  _Float16 v8h;
typedef __attribute__((ext_vector_type(8)))  float    v8f;

__device__ __forceinline__ v8f wmma_f32_f16(v16h a, v16h b, v8f c) {
    // D = A(16x32 f16) * B(32x16 f16) + C(16x16 f32)
    return __builtin_amdgcn_wmma_f32_16x16x32_f16(false, a, false, b, (short)0, c, false, false);
}

// ---------------- prep kernels ----------------

__global__ void k_cvt_pad_x(const float* __restrict__ x, _Float16* __restrict__ xh) {
    int idx = blockIdx.x * blockDim.x + threadIdx.x;   // 512*1024 total
    int m = idx >> 10, k = idx & 1023;
    xh[idx] = (m < 500) ? (_Float16)x[m * 1024 + k] : (_Float16)0.0f;
}

// src: K x N (row-major f32)  ->  dst: N x K (row-major f16)
__global__ void k_transpose_cvt(const float* __restrict__ src, _Float16* __restrict__ dst,
                                int K, int N) {
    int idx = blockIdx.x * blockDim.x + threadIdx.x;
    if (idx >= K * N) return;
    int n = idx / K, k = idx - n * K;
    dst[idx] = (_Float16)src[k * N + n];
}

// hbar[I][c] = mean over 5 rows of h
__global__ void k_pool(const float* __restrict__ H, float* __restrict__ hbar) {
    int I = blockIdx.x, c = threadIdx.x;               // 100 blocks x 512 threads
    const float* r = H + (size_t)I * 5 * 512 + c;
    hbar[I * 512 + c] = (r[0] + r[512] + r[1024] + r[1536] + r[2048]) * 0.2f;
}

// ---------------- generic WMMA GEMM: out = act(A @ B + bias) ----------------
// A: M x K f16 row-major. BT: N x K f16 (B transposed). 128x128 tile / block.
template <bool RELU, bool OUT_F16>
__global__ void __launch_bounds__(256, 1)
k_gemm_wmma(const _Float16* __restrict__ A,
            const _Float16* __restrict__ BT,
            const float* __restrict__ bias,
            void* __restrict__ out,
            int K, int N) {
    __shared__ __align__(32) _Float16 sA[8 * 512];     // 8 M-tiles, frag layout, 8 KB
    const int tid  = threadIdx.x;
    const int wave = tid >> 5, lane = tid & 31;
    const int wm = wave & 1, wn = wave >> 1;           // 2 x 4 wave grid
    const int m0 = blockIdx.x * 128;
    const int nw = blockIdx.y * 128 + wn * 32;
    v8f acc[4][2] = {};
    for (int k0 = 0; k0 < K; k0 += 32) {
        {   // stage A chunk (128 x 32) into fragment layout
            const int mloc = tid >> 1, hh = tid & 1;
            const _Float16* src = A + (size_t)(m0 + mloc) * K + k0 + 16 * hh;
            v8h lo = *(const v8h*)src;
            v8h hi = *(const v8h*)(src + 8);
            _Float16* d = sA + (mloc >> 4) * 512 + ((mloc & 15) + 16 * hh) * 16;
            *(v8h*)d = lo;
            *(v8h*)(d + 8) = hi;
        }
        __syncthreads();
#pragma unroll
        for (int nt = 0; nt < 2; ++nt) {
            const int nn = nw + nt * 16 + (lane & 15);
            v16h bf = *(const v16h*)(BT + (size_t)nn * K + k0 + 16 * (lane >> 4));
#pragma unroll
            for (int mt = 0; mt < 4; ++mt) {
                v16h af = *(const v16h*)(sA + (wm * 4 + mt) * 512 + lane * 16);
                acc[mt][nt] = wmma_f32_f16(af, bf, acc[mt][nt]);
            }
        }
        __syncthreads();
    }
#pragma unroll
    for (int nt = 0; nt < 2; ++nt) {
        const int nn = nw + nt * 16 + (lane & 15);
        const float bb = bias[nn];
#pragma unroll
        for (int mt = 0; mt < 4; ++mt) {
            const int mb = m0 + (wm * 4 + mt) * 16 + 8 * (lane >> 4);
#pragma unroll
            for (int v = 0; v < 8; ++v) {
                float val = acc[mt][nt][v] + bb;
                if (RELU) val = fmaxf(val, 0.0f);
                if (OUT_F16) ((_Float16*)out)[(size_t)(mb + v) * N + nn] = (_Float16)val;
                else         ((float*)out)[(size_t)(mb + v) * N + nn] = val;
            }
        }
    }
}

// ---------------- fused pair->GEMM kernel ----------------
// out[i][j][n] = sum_c (0.5*(H[i][c]+H[j][c]) + H[i][c]*H[j][c]/sqrt(512)) * B[c][n] + bias[n]
// Block: one 16x16 (i,j) tile, 8 waves each own a 16-wide N strip (N=128), K=512.
__global__ void __launch_bounds__(256, 1)
k_pair_gemm_wmma(const float* __restrict__ H,      // n(+pad) x 512 f32
                 const _Float16* __restrict__ BT,  // 128 x 512 f16
                 const float* __restrict__ bias,   // 128
                 float* __restrict__ out,          // n x n x 128 f32
                 int n) {
    __shared__ __align__(32) _Float16 sP[16 * 512];    // 16 M-tiles, 16 KB
    const float SC = 0.04419417382415922f;             // 1/sqrt(512)
    const int tid  = threadIdx.x;
    const int wave = tid >> 5, lane = tid & 31;
    const int i0 = blockIdx.y * 16, j0 = blockIdx.x * 16;
    __builtin_amdgcn_s_wait_tensorcnt(0);
    __builtin_prefetch(BT, 0, 1);
    v8f acc[16] = {};
    for (int k0 = 0; k0 < 512; k0 += 32) {
        {   // compute pair chunk (256 rows x 32 K) into fragment layout
            const int ii = tid >> 4, jj = tid & 15;
            const int i = i0 + ii, j = j0 + jj;
            const bool ok = (i < n) && (j < n);
            _Float16* d = sP + ii * 512 + jj * 16;
#pragma unroll
            for (int hh = 0; hh < 2; ++hh) {
                float p0[8], p1[8];
                if (ok) {
                    const float* hi = H + (size_t)i * 512 + k0 + 16 * hh;
                    const float* hj = H + (size_t)j * 512 + k0 + 16 * hh;
                    v8f a0 = *(const v8f*)hi, a1 = *(const v8f*)(hi + 8);
                    v8f b0 = *(const v8f*)hj, b1 = *(const v8f*)(hj + 8);
#pragma unroll
                    for (int e = 0; e < 8; ++e) {
                        p0[e] = 0.5f * (a0[e] + b0[e]) + SC * a0[e] * b0[e];
                        p1[e] = 0.5f * (a1[e] + b1[e]) + SC * a1[e] * b1[e];
                    }
                } else {
#pragma unroll
                    for (int e = 0; e < 8; ++e) { p0[e] = 0.0f; p1[e] = 0.0f; }
                }
                v8h o0, o1;
#pragma unroll
                for (int e = 0; e < 8; ++e) { o0[e] = (_Float16)p0[e]; o1[e] = (_Float16)p1[e]; }
                *(v8h*)(d + hh * 256)     = o0;
                *(v8h*)(d + hh * 256 + 8) = o1;
            }
        }
        __syncthreads();
        const int nn = wave * 16 + (lane & 15);
        v16h bf = *(const v16h*)(BT + (size_t)nn * 512 + k0 + 16 * (lane >> 4));
#pragma unroll
        for (int mt = 0; mt < 16; ++mt) {
            v16h af = *(const v16h*)(sP + mt * 512 + lane * 16);
            acc[mt] = wmma_f32_f16(af, bf, acc[mt]);
        }
        __syncthreads();
    }
    const int nn = wave * 16 + (lane & 15);
    const float bb = bias[nn];
#pragma unroll
    for (int mt = 0; mt < 16; ++mt) {
        const int i = i0 + mt;
        if (i >= n) continue;
#pragma unroll
        for (int v = 0; v < 8; ++v) {
            const int j = j0 + v + 8 * (lane >> 4);
            if (j < n) out[((size_t)i * n + j) * 128 + nn] = acc[mt][v] + bb;
        }
    }
}

// ---------------- hidden = relu(mc @ ih_w1 + b1) via WMMA ----------------
// mc rows: r=a*400+b -> micro[(a+50)*500 + (b+50)]. M=160000, K=128, N=64.
__global__ void __launch_bounds__(256, 1)
k_hidden_wmma(const float* __restrict__ micro,     // 500x500x128 f32
              const _Float16* __restrict__ BT,     // 64 x 128 f16
              const float* __restrict__ bias,      // 64
              _Float16* __restrict__ hidden) {     // 160000 x 64 f16
    __shared__ __align__(32) _Float16 sA[16 * 512];
    const int tid  = threadIdx.x;
    const int wave = tid >> 5, lane = tid & 31;
    const int wn = wave & 3, wmh = wave >> 2;          // 4 N-subtiles x 2 M-halves
    const int r0 = blockIdx.x * 256;
    v8f acc[8] = {};
    for (int k0 = 0; k0 < 128; k0 += 32) {
        {
            const int r = r0 + tid;
            const int a = r / 400, b = r - a * 400;
            const float* src = micro + ((size_t)(a + 50) * 500 + (b + 50)) * 128 + k0;
            _Float16* d = sA + (tid >> 4) * 512 + (tid & 15) * 16;
#pragma unroll
            for (int hh = 0; hh < 2; ++hh) {
                v8f a0 = *(const v8f*)(src + 16 * hh);
                v8f a1 = *(const v8f*)(src + 16 * hh + 8);
                v8h o0, o1;
#pragma unroll
                for (int e = 0; e < 8; ++e) { o0[e] = (_Float16)a0[e]; o1[e] = (_Float16)a1[e]; }
                *(v8h*)(d + hh * 256)     = o0;
                *(v8h*)(d + hh * 256 + 8) = o1;
            }
        }
        __syncthreads();
        const int nn = wn * 16 + (lane & 15);
        v16h bf = *(const v16h*)(BT + (size_t)nn * 128 + k0 + 16 * (lane >> 4));
#pragma unroll
        for (int t = 0; t < 8; ++t) {
            v16h af = *(const v16h*)(sA + (wmh * 8 + t) * 512 + lane * 16);
            acc[t] = wmma_f32_f16(af, bf, acc[t]);
        }
        __syncthreads();
    }
    const int nn = wn * 16 + (lane & 15);
    const float bb = bias[nn];
#pragma unroll
    for (int t = 0; t < 8; ++t) {
        const int rb = r0 + (wmh * 8 + t) * 16 + 8 * (lane >> 4);
#pragma unroll
        for (int v = 0; v < 8; ++v) {
            float val = fmaxf(acc[t][v] + bb, 0.0f);
            hidden[(size_t)(rb + v) * 64 + nn] = (_Float16)val;
        }
    }
}

// ---------------- small heads (VALU dots) ----------------

__global__ void k_heads_mc_ih(const float* __restrict__ micro,
                              const _Float16* __restrict__ hidden,
                              const float* __restrict__ mc_w, const float* __restrict__ mc_b,
                              const float* __restrict__ ih_w2, const float* __restrict__ ih_b2,
                              float* __restrict__ out_mc, float* __restrict__ out_ih) {
    int r = blockIdx.x * 256 + threadIdx.x;            // < 160000
    int a = r / 400, b = r - a * 400;
    const float* mc = micro + ((size_t)(a + 50) * 500 + (b + 50)) * 128;
    float s0 = mc_b[0], s1 = mc_b[1];
    for (int k = 0; k < 128; ++k) {
        float v = mc[k];
        s0 = fmaf(v, mc_w[2 * k], s0);
        s1 = fmaf(v, mc_w[2 * k + 1], s1);
    }
    out_mc[2 * r] = s0; out_mc[2 * r + 1] = s1;
    const _Float16* hd = hidden + (size_t)r * 64;
    float t0 = ih_b2[0], t1 = ih_b2[1];
    for (int k = 0; k < 64; ++k) {
        float v = (float)hd[k];
        t0 = fmaf(v, ih_w2[2 * k], t0);
        t1 = fmaf(v, ih_w2[2 * k + 1], t1);
    }
    out_ih[2 * r] = t0; out_ih[2 * r + 1] = t1;
}

__global__ void k_head_hic(const float* __restrict__ hic,
                           const float* __restrict__ w, const float* __restrict__ b,
                           float* __restrict__ out) {
    int t = blockIdx.x * 256 + threadIdx.x;            // < 6400
    int ib = t / 80, jb = t - ib * 80;
    const float* row = hic + ((size_t)(ib + 10) * 100 + (jb + 10)) * 128;
    float s0 = b[0], s1 = b[1], s2 = b[2];
    for (int k = 0; k < 128; ++k) {
        float v = row[k];
        s0 = fmaf(v, w[3 * k], s0);
        s1 = fmaf(v, w[3 * k + 1], s1);
        s2 = fmaf(v, w[3 * k + 2], s2);
    }
    out[3 * t] = s0; out[3 * t + 1] = s1; out[3 * t + 2] = s2;
}

// ---------------- launch ----------------

extern "C" void kernel_launch(void* const* d_in, const int* in_sizes, int n_in,
                              void* d_out, int out_size, void* d_ws, size_t ws_size,
                              hipStream_t stream) {
    const float* x       = (const float*)d_in[0];
    const float* p_w1    = (const float*)d_in[1];
    const float* p_b1    = (const float*)d_in[2];
    const float* p_w2    = (const float*)d_in[3];
    const float* p_b2    = (const float*)d_in[4];
    const float* micro_w = (const float*)d_in[5];
    const float* micro_b = (const float*)d_in[6];
    const float* hic_w   = (const float*)d_in[7];
    const float* hic_b   = (const float*)d_in[8];
    const float* mc_w    = (const float*)d_in[9];
    const float* mc_b    = (const float*)d_in[10];
    const float* ih_w1   = (const float*)d_in[11];
    const float* ih_b1   = (const float*)d_in[12];
    const float* ih_w2   = (const float*)d_in[13];
    const float* ih_b2   = (const float*)d_in[14];
    const float* hic_hw  = (const float*)d_in[15];
    const float* hic_hb  = (const float*)d_in[16];

    char* ws = (char*)d_ws;
    _Float16* xh     = (_Float16*)(ws + 0);            // 512x1024 f16
    _Float16* w1T    = (_Float16*)(ws + 1048576);      // 512x1024 f16
    _Float16* h1h    = (_Float16*)(ws + 2097152);      // 512x512  f16
    _Float16* w2T    = (_Float16*)(ws + 2621440);      // 512x512  f16
    float*    hf     = (float*)   (ws + 3145728);      // 512x512  f32
    _Float16* microT = (_Float16*)(ws + 4194304);      // 128x512  f16
    _Float16* hicT   = (_Float16*)(ws + 4325376);      // 128x512  f16
    _Float16* ihw1T  = (_Float16*)(ws + 4456448);      // 64x128   f16
    float*    hbar   = (float*)   (ws + 4472832);      // 100x512  f32
    float*    hic    = (float*)   (ws + 4677632);      // 100x100x128 f32
    _Float16* hidden = (_Float16*)(ws + 9797632);      // 160000x64 f16

    float* out       = (float*)d_out;
    float* out_mc    = out;                            // 400*400*2
    float* out_ih    = out + 320000;                   // 400*400*2
    float* out_micro = out + 640000;                   // 500*500*128
    float* out_hic   = out + 32640000;                 // 80*80*3

    k_cvt_pad_x<<<2048, 256, 0, stream>>>(x, xh);
    k_transpose_cvt<<<2048, 256, 0, stream>>>(p_w1, w1T, 1024, 512);
    k_transpose_cvt<<<1024, 256, 0, stream>>>(p_w2, w2T, 512, 512);
    k_transpose_cvt<<<256,  256, 0, stream>>>(micro_w, microT, 512, 128);
    k_transpose_cvt<<<256,  256, 0, stream>>>(hic_w, hicT, 512, 128);
    k_transpose_cvt<<<32,   256, 0, stream>>>(ih_w1, ihw1T, 128, 64);

    k_gemm_wmma<true,  true ><<<dim3(4, 4), 256, 0, stream>>>(xh,  w1T, p_b1, (void*)h1h, 1024, 512);
    k_gemm_wmma<false, false><<<dim3(4, 4), 256, 0, stream>>>(h1h, w2T, p_b2, (void*)hf,  512,  512);

    k_pool<<<100, 512, 0, stream>>>(hf, hbar);

    k_pair_gemm_wmma<<<dim3(32, 32), 256, 0, stream>>>(hf,   microT, micro_b, out_micro, 500);
    k_pair_gemm_wmma<<<dim3(7, 7),   256, 0, stream>>>(hbar, hicT,   hic_b,   hic,       100);

    k_head_hic<<<25, 256, 0, stream>>>(hic, hic_hw, hic_hb, out_hic);
    k_hidden_wmma<<<625, 256, 0, stream>>>(out_micro, ihw1T, ih_b1, hidden);
    k_heads_mc_ih<<<625, 256, 0, stream>>>(out_micro, hidden, mc_w, mc_b, ih_w2, ih_b2, out_mc, out_ih);
}
